// GcnPolicy_65051574665497
// MI455X (gfx1250) — matrix operand order repaired and checked by reading, not compiled, not run on hardware
//
#include <hip/hip_runtime.h>
#include <hip/hip_bf16.h>

#define NB 8192
#define NN 64
#define MM 64
#define NTHREADS 512
#define ACTION_COLS 193
#define ACTION_TOTAL (NB * ACTION_COLS)

typedef __attribute__((ext_vector_type(16))) _Float16 v16h;
typedef __attribute__((ext_vector_type(8)))  _Float16 v8h;
typedef __attribute__((ext_vector_type(8)))  float    v8f;
typedef __attribute__((ext_vector_type(4)))  unsigned int v4u;
typedef __attribute__((ext_vector_type(4)))  int v4i;
typedef __attribute__((ext_vector_type(8)))  int v8i;

#if __has_builtin(__builtin_amdgcn_tensor_load_to_lds) && \
    __has_builtin(__builtin_amdgcn_s_wait_tensorcnt)
#define USE_TDM 1
#endif

// ---- WMMA fragment helpers (layouts per CDNA5 ISA 7.12.2, wave32) ----
// All operands are laid out so every lane's 16 elements are two K-runs of 8 (table order):
// two ds_load_b128 per fragment, one ds_store_b128 per D-tile.

// A fragment: row-major M x K matrix (stride 64 halves), tile at (row0, k0).
__device__ inline v16h frag_a(const _Float16* m, int row0, int k0, int lane) {
  const int r  = row0 + (lane & 15);
  const int hk = ((lane >> 4) & 1) << 3;   // K sub-offset 0 or 8
  const _Float16* p = m + r * 64 + k0 + hk;
  v8h lo = *reinterpret_cast<const v8h*>(p);        // K = k0+hk    .. +7
  v8h hi = *reinterpret_cast<const v8h*>(p + 16);   // K = k0+16+hk .. +7
  return __builtin_shufflevector(lo, hi, 0, 1, 2, 3, 4, 5, 6, 7,
                                         8, 9, 10, 11, 12, 13, 14, 15);
}

// B fragment: K x N matrix stored TRANSPOSED as XT[n*64 + k].
// lane holds column col0 + lane%16; K half picked by lane>=16; two K runs of 8, 16 apart.
__device__ inline v16h frag_bT(const _Float16* xt, int k0, int col0, int lane) {
  const int n  = col0 + (lane & 15);
  const int kb = k0 + (((lane >> 4) & 1) << 3);     // 0 or 8
  const _Float16* p = xt + n * 64 + kb;
  v8h lo = *reinterpret_cast<const v8h*>(p);        // K = kb    .. +7
  v8h hi = *reinterpret_cast<const v8h*>(p + 16);   // K = kb+16 .. +7
  return __builtin_shufflevector(lo, hi, 0, 1, 2, 3, 4, 5, 6, 7,
                                         8, 9, 10, 11, 12, 13, 14, 15);
}

// D store TRANSPOSED: element (m,n) -> dt[n*64 + m]; lane's 8 rows are
// contiguous -> one packed 16B store. Optional ReLU.
__device__ inline void store_frag_T(_Float16* dt, int row0, int col0, int lane,
                                    v8f acc, bool do_relu) {
  const int n  = col0 + (lane & 15);
  const int r0 = row0 + (((lane >> 4) & 1) << 3);
  v8h pk;
#pragma unroll
  for (int r = 0; r < 8; ++r) {
    float v = acc[r];
    if (do_relu) v = fmaxf(v, 0.0f);
    pk[r] = (_Float16)v;
  }
  *reinterpret_cast<v8h*>(dt + n * 64 + r0) = pk;
}

// 64x64x64 matmul: A row-major, B transposed-stored, D stored transposed.
// 16 waves, one 16x16 tile each, K = 2 x 32.
__device__ inline void mm64(const _Float16* A_, const _Float16* BT_,
                            _Float16* DT_, int lane, int wave, bool do_relu) {
  const int mt = (wave >> 2) << 4;
  const int nt = (wave & 3) << 4;
  v8f acc = {};
  v16h a0 = frag_a(A_, mt, 0, lane);
  v16h b0 = frag_bT(BT_, 0, nt, lane);
  acc = __builtin_amdgcn_wmma_f32_16x16x32_f16(false, a0, false, b0,
                                               (short)0, acc, false, false);
  v16h a1 = frag_a(A_, mt, 32, lane);
  v16h b1 = frag_bT(BT_, 32, nt, lane);
  acc = __builtin_amdgcn_wmma_f32_16x16x32_f16(false, a1, false, b1,
                                               (short)0, acc, false, false);
  store_frag_T(DT_, mt, nt, lane, acc, do_relu);
}

// Zero-padded TRANSPOSED weight upload: R x C f32 weight -> WT[c*64 + r] f16.
__device__ inline void load_weight_padT(_Float16* sWT, const float* W,
                                        int R, int C, int tid) {
  for (int i = tid; i < 64 * 64; i += NTHREADS) {
    const int c = i >> 6, r = i & 63;               // i == c*64 + r
    float v = (r < R && c < C) ? W[r * C + c] : 0.0f;
    sWT[i] = (_Float16)v;
  }
}

__device__ inline float softmax64(const float* v, int idx) {
  float mx = v[0];
#pragma unroll 8
  for (int i = 1; i < 64; ++i) mx = fmaxf(mx, v[i]);
  float s = 0.0f;
#pragma unroll 8
  for (int i = 0; i < 64; ++i) s += __expf(v[i] - mx);
  return __expf(v[idx] - mx) / s;
}

__global__ __launch_bounds__(NTHREADS) void gcn_policy_kernel(
    const float* __restrict__ adj,
    const float* __restrict__ W1, const float* __restrict__ W2,
    const float* __restrict__ W3, const float* __restrict__ Wc1,
    const float* __restrict__ Wc2, const float* __restrict__ Wc3,
    const float* __restrict__ mf_w, const float* __restrict__ mf_b,
    const float* __restrict__ ms_w, const float* __restrict__ ms_b,
    const float* __restrict__ me_w, const float* __restrict__ me_b,
    const float* __restrict__ cr_w, const float* __restrict__ cr_b,
    float* __restrict__ out) {
  __shared__ __attribute__((aligned(16))) float    sA[NN * NN];     // raw A f32
  __shared__ __attribute__((aligned(16))) _Float16 sNorm[NN * NN];  // normA row-major
  __shared__ __attribute__((aligned(16))) _Float16 sHT[NN * NN];    // h transposed
  __shared__ __attribute__((aligned(16))) _Float16 sT[NN * NN];     // T row-major
  __shared__ __attribute__((aligned(16))) _Float16 sWT[NN * NN];    // weight transposed
  __shared__ float sD[NN];
  __shared__ float sXs[8];
  __shared__ float sXf[MM], sXsec[MM], sXedge[MM];

  const int tid  = threadIdx.x;
  const int lane = tid & 31;
  const int wave = tid >> 5;
  const int b    = blockIdx.x;
  const float* Ab = adj + (size_t)b * NN * NN;      // adj[b,0,:,:]

  // ---- stage adjacency tile into LDS ----
#ifdef USE_TDM
  if (wave == 0) {
    // Tensor DMA descriptor (ISA 8.3/8.4): 2D, 64x64, 4-byte elements.
    const unsigned long long ga = (unsigned long long)(uintptr_t)Ab;
    const unsigned int ldsb = (unsigned int)(uintptr_t)&sA[0];
    v4u g0 = { 1u,                                   // count=1, user mode
               ldsb,                                 // lds_addr (bytes)
               (unsigned int)ga,                     // global_addr[31:0]
               ((unsigned int)(ga >> 32) & 0x01FFFFFFu) | 0x80000000u }; // [56:32]|type=2
    v8i g1 = { (int)0x00020000,                      // data_size=2 (4B), mask=0
               (int)0x00400000,                      // tensor_dim0=64 (bit48)
               (int)0x00400000,                      // tensor_dim1=64 (bit80)
               (int)0x00400000,                      // tile_dim0=64  (bit112)
               64,                                   // tile_dim1=64  (bit128)
               64,                                   // dim0_stride=64 (bit160)
               0, 0 };
    v4i gz = {0, 0, 0, 0};
#if defined(__clang_major__) && __clang_major__ >= 23
    v8i gz8 = {0, 0, 0, 0, 0, 0, 0, 0};
    __builtin_amdgcn_tensor_load_to_lds(g0, g1, gz, gz, gz8, 0);
#else
    __builtin_amdgcn_tensor_load_to_lds(g0, g1, gz, gz, 0);
#endif
    __builtin_amdgcn_s_wait_tensorcnt(0);
  }
  __syncthreads();
#else
  {
    const float4* src = reinterpret_cast<const float4*>(Ab);
    float4* dst = reinterpret_cast<float4*>(sA);
    for (int i = tid; i < (NN * NN) / 4; i += NTHREADS) dst[i] = src[i];
  }
  __syncthreads();
#endif

  // ---- degree (column sums) and inverse-sqrt ----
  if (tid < NN) {
    float s = 0.0f;
#pragma unroll 8
    for (int i = 0; i < NN; ++i) s += sA[i * NN + tid];
    sD[tid] = rsqrtf(s + 1.0f);
  }
  __syncthreads();

  // ---- normA = d_i * (A + I) * d_j  (f16, row-major) ----
  for (int i = tid; i < NN * NN; i += NTHREADS) {
    const int r = i >> 6, c = i & 63;
    float ah = sA[i] + ((r == c) ? 1.0f : 0.0f);
    sNorm[i] = (_Float16)(sD[r] * ah * sD[c]);
  }
  __syncthreads();

  const float* Wact[3]  = {W1, W2, W3};
  const float* Wcri[3]  = {Wc1, Wc2, Wc3};
  const int    Wrows[3] = {64, 20, 16};
  const int    Wcols[3] = {20, 16, 8};

  for (int pass = 0; pass < 2; ++pass) {
    // h0^T = A^T  (f16)
    for (int i = tid; i < NN * NN; i += NTHREADS) {
      const int f = i >> 6, n = i & 63;
      sHT[i] = (_Float16)sA[n * 64 + f];
    }
    __syncthreads();

    const float* const* Wset = (pass == 0) ? Wact : Wcri;
#pragma unroll 1
    for (int l = 0; l < 3; ++l) {
      // stage transposed padded weight; independent of aggregation buffers
      load_weight_padT(sWT, Wset[l], Wrows[l], Wcols[l], tid);
      // T^T = h^T @ normA^T  (B-operand normA^T is "transposed-stored" == sNorm
      //  row-major); D stored transposed -> sT holds T row-major.
      mm64(sHT, sNorm, sT, lane, wave, false);
      __syncthreads();
      // h' = relu(T @ Wpad); D stored transposed -> sHT holds h'^T.
      mm64(sT, sWT, sHT, lane, wave, true);
      __syncthreads();
    }

    if (pass == 0) {
      // ---- Xs[c] = sum over nodes of X[:,c]  (contiguous rows of sHT) ----
      if (tid < 8) {
        float s = 0.0f;
#pragma unroll 8
        for (int i = 0; i < NN; ++i) s += (float)sHT[tid * 64 + i];
        sXs[tid] = s;
      }
      __syncthreads();
      // ---- heads ----
      if (tid < MM) {
        float v = mf_b[tid];
#pragma unroll
        for (int k = 0; k < 8; ++k) v += sXs[k] * mf_w[k * MM + tid];
        sXf[tid] = v;
      }
      __syncthreads();
      if (tid < MM) {
        float v = ms_b[tid];
#pragma unroll 8
        for (int k = 0; k < MM; ++k) v += sXf[k] * ms_w[k * MM + tid];
#pragma unroll
        for (int k = 0; k < 8; ++k) v += sXs[k] * ms_w[(MM + k) * MM + tid];
        sXsec[tid] = v;
      }
      __syncthreads();
      if (tid < MM) {
        float v = me_b[tid];
#pragma unroll 8
        for (int k = 0; k < MM; ++k) v += sXf[k] * me_w[k * MM + tid];
#pragma unroll 8
        for (int k = 0; k < MM; ++k) v += sXsec[k] * me_w[(MM + k) * MM + tid];
        sXedge[tid] = v;
      }
      __syncthreads();
      // ---- softmaxes + action output ----
      if (tid < MM) {
        float* arow = out + (size_t)b * ACTION_COLS;
        arow[tid]          = softmax64(sXf, tid);
        arow[MM + tid]     = softmax64(sXsec, tid);
        arow[2 * MM + tid] = softmax64(sXedge, tid);
      }
      if (tid == 0) out[(size_t)b * ACTION_COLS + 192] = 1.0f; // softmax(scalar)
      __syncthreads();
    } else {
      // ---- critic[i] = X[i,:] . cr_w + cr_b ----
      if (tid < NN) {
        float v = cr_b[0];
#pragma unroll
        for (int k = 0; k < 8; ++k) v += (float)sHT[k * 64 + tid] * cr_w[k];
        out[(size_t)ACTION_TOTAL + (size_t)b * NN + tid] = v;
      }
    }
  }
}

extern "C" void kernel_launch(void* const* d_in, const int* in_sizes, int n_in,
                              void* d_out, int out_size, void* d_ws, size_t ws_size,
                              hipStream_t stream) {
  (void)in_sizes; (void)n_in; (void)out_size; (void)d_ws; (void)ws_size;
  const float* adj  = (const float*)d_in[0];
  const float* W1   = (const float*)d_in[1];
  const float* W2   = (const float*)d_in[2];
  const float* W3   = (const float*)d_in[3];
  const float* Wc1  = (const float*)d_in[4];
  const float* Wc2  = (const float*)d_in[5];
  const float* Wc3  = (const float*)d_in[6];
  const float* mf_w = (const float*)d_in[7];
  const float* mf_b = (const float*)d_in[8];
  const float* ms_w = (const float*)d_in[9];
  const float* ms_b = (const float*)d_in[10];
  const float* me_w = (const float*)d_in[11];
  const float* me_b = (const float*)d_in[12];
  // d_in[13]=mt_w, d_in[14]=mt_b unused: softmax over one element == 1.0
  const float* cr_w = (const float*)d_in[15];
  const float* cr_b = (const float*)d_in[16];
  float* out = (float*)d_out;

  gcn_policy_kernel<<<NB, NTHREADS, 0, stream>>>(
      adj, W1, W2, W3, Wc1, Wc2, Wc3, mf_w, mf_b, ms_w, ms_b, me_w, me_b,
      cr_w, cr_b, out);
}
